// Kernel_transformer_39273180954985
// MI455X (gfx1250) — matrix operand order repaired and checked by reading, not compiled, not run on hardware
//
#include <hip/hip_runtime.h>
#include <hip/hip_bf16.h>

// ---------------- problem constants ----------------
#define B_   8
#define S_   8192
#define D_   256
#define H_   8
#define L_   4
#define F_   1024
#define DH_  32
#define M_TOK (B_ * S_)          // 65536 tokens

// ---------------- WMMA types ----------------
typedef __attribute__((ext_vector_type(16))) __bf16 v16bf;
typedef __attribute__((ext_vector_type(8)))  float  v8f;

// ---------------- bf16 <-> f32 helpers (storage = unsigned short) -------------
__device__ __forceinline__ unsigned short f2bf(float f) {
    union { float f; unsigned int u; } x; x.f = f;
    unsigned int u = x.u + 0x7FFFu + ((x.u >> 16) & 1u);   // RNE
    return (unsigned short)(u >> 16);
}
__device__ __forceinline__ float bf2f(unsigned short h) {
    union { unsigned int u; float f; } x; x.u = ((unsigned int)h) << 16;
    return x.f;
}

// =================================================================
// Generic bf16 WMMA GEMM:  C[M,N] = A[M,K] * Bw[K,N]   (row-major)
//   relu     : apply max(0,·) to the accumulator
//   rowscale : optional per-row multiplier (key/value padding mask)
//   resid    : if non-null -> resid[idx] += alpha*acc ; Cout = bf16(resid)
//              else        -> Cout = bf16(acc)
//   alpha    : read on-device from alphap[alphaidx] (ReZero scalar)
// Block tile 64x256, 8 waves (2x4), each wave computes 32x64
// (2 A-frags x 4 B-frags -> 8 WMMAs per K-step).
// LDS double-buffered: stage tile k+1 while WMMA consumes tile k.
// Requires M%64==0, N%256==0, K%32==0 (true for all uses here).
// =================================================================
#define BM 64
#define BN 256
#define BK 32
#define LDA 40   // LDS pitch (elements) for A tile rows  (16B-aligned, bank-skewed)
#define LDB 40   // LDS pitch for transposed B tile rows

__global__ void __launch_bounds__(256)
gemm_bf16(const unsigned short* __restrict__ A,
          const unsigned short* __restrict__ Bw,
          int M, int N, int K, int relu,
          const float* __restrict__ rowscale,
          unsigned short* __restrict__ Cout,
          float* __restrict__ resid,
          const float* __restrict__ alphap, int alphaidx)
{
    __shared__ unsigned short As[2][BM * LDA];   // 2 x  5.0 KB
    __shared__ unsigned short Bs[2][BN * LDB];   // 2 x 20.0 KB

    const int tid  = threadIdx.x;
    const int lane = tid & 31;
    const int wid  = tid >> 5;
    const int half = lane >> 4;
    const int lm   = lane & 15;

    const int m0 = blockIdx.y * BM;
    const int n0 = blockIdx.x * BN;
    const int waveM = (wid >> 2) * 32;   // 0 or 32
    const int waveN = (wid & 3) * 64;    // 0,64,128,192

    v8f acc[2][4] = {};

    auto stage = [&](int buf, int k0) {
        // ---- A tile (64 x 32), row-major, vectorized 4 x bf16 ----
#pragma unroll
        for (int t = 0; t < (BM * BK) / (4 * 256); ++t) {   // 2 iters
            int idx = (t * 256 + tid) * 4;
            int r = idx >> 5;          // /32
            int c = idx & 31;
            uint2 d = *(const uint2*)(A + (size_t)(m0 + r) * K + k0 + c);
            *(uint2*)&As[buf][r * LDA + c] = d;
        }
        // ---- B tile (32 x 256) transposed -> Bs[n][k] ----
#pragma unroll
        for (int t = 0; t < (BK * BN) / (4 * 256); ++t) {   // 8 iters
            int idx = (t * 256 + tid) * 4;
            int kr = idx >> 8;         // /256
            int nc = idx & 255;
            uint2 d = *(const uint2*)(Bw + (size_t)(k0 + kr) * N + n0 + nc);
            const unsigned short* e = (const unsigned short*)&d;
            Bs[buf][(nc + 0) * LDB + kr] = e[0];
            Bs[buf][(nc + 1) * LDB + kr] = e[1];
            Bs[buf][(nc + 2) * LDB + kr] = e[2];
            Bs[buf][(nc + 3) * LDB + kr] = e[3];
        }
    };

    stage(0, 0);
    __syncthreads();

    const int Ktiles = K / BK;
    for (int kt = 0; kt < Ktiles; ++kt) {
        const int cur = kt & 1;
        if (kt + 1 < Ktiles) stage(cur ^ 1, (kt + 1) * BK);

        // ---- build fragments per ISA 16-bit layouts ----
        // A 16x32: lanes0-15 elems0-7=K0-7, elems8-15=K16-23; lanes16-31 +8
        v16bf afr[2], bfr[4];
#pragma unroll
        for (int i = 0; i < 2; ++i) {
            int m = waveM + i * 16 + lm;
            int off = m * LDA + 8 * half;
            uint4* dst = reinterpret_cast<uint4*>(&afr[i]);
            dst[0] = *reinterpret_cast<const uint4*>(&As[cur][off]);
            dst[1] = *reinterpret_cast<const uint4*>(&As[cur][off + 16]);
        }
        // B 32x16: col = lane%16; lanes0-15 elems0-7=K0-7, elems8-15=K16-23; lanes16-31 +8
#pragma unroll
        for (int j = 0; j < 4; ++j) {
            int n = waveN + j * 16 + lm;
            int off = n * LDB + 8 * half;
            uint4* dst = reinterpret_cast<uint4*>(&bfr[j]);
            const uint4* src = reinterpret_cast<const uint4*>(&Bs[cur][off]);
            dst[0] = src[0];
            dst[1] = src[2];
        }
#pragma unroll
        for (int i = 0; i < 2; ++i)
#pragma unroll
            for (int j = 0; j < 4; ++j)
                acc[i][j] = __builtin_amdgcn_wmma_f32_16x16x32_bf16(
                    false, afr[i], false, bfr[j], (short)0, acc[i][j], false, false);
        __syncthreads();
    }

    // ---- epilogue ----
    const float alpha = alphap ? alphap[alphaidx] : 1.0f;
#pragma unroll
    for (int i = 0; i < 2; ++i) {
#pragma unroll
        for (int j = 0; j < 4; ++j) {
            int rbase = m0 + waveM + i * 16 + 8 * half;
            int col   = n0 + waveN + j * 16 + lm;
#pragma unroll
            for (int r = 0; r < 8; ++r) {
                int row = rbase + r;
                float v = acc[i][j][r];
                if (relu) v = v > 0.f ? v : 0.f;
                if (rowscale) v *= rowscale[row];
                size_t idx = (size_t)row * N + col;
                if (resid) {
                    float nv = resid[idx] + alpha * v;
                    resid[idx] = nv;
                    Cout[idx] = f2bf(nv);
                } else {
                    Cout[idx] = f2bf(v);
                }
            }
        }
    }
}

// =================================================================
// kv_summary: per (b,h) compute
//   kv_c[d,e] = sum_s k[s,d]*cos(s) * v[s,e],  kv_s likewise with sin
//   ksc[d] = sum_s k[s,d]*cos(s),  kss[d] with sin
// One workgroup per (b,h); 256 threads; each thread owns 4 (d,e) pairs.
// =================================================================
#define KVCHUNK 64
__global__ void __launch_bounds__(256)
kv_summary(const unsigned short* __restrict__ Kb,
           const unsigned short* __restrict__ Vb,
           const int* __restrict__ lengths,
           float* __restrict__ kvc, float* __restrict__ kvs,
           float* __restrict__ ksc, float* __restrict__ kss)
{
    const int b = blockIdx.x >> 3;
    const int h = blockIdx.x & 7;
    const int tid = threadIdx.x;

    __shared__ unsigned short Ks[KVCHUNK][DH_];
    __shared__ unsigned short Vs[KVCHUNK][DH_];
    __shared__ float Cs[KVCHUNK], Ss[KVCHUNK];

    float accc[4] = {0.f, 0.f, 0.f, 0.f};
    float accs[4] = {0.f, 0.f, 0.f, 0.f};
    float kc_sum[4] = {0.f, 0.f, 0.f, 0.f};
    float ks_sum[4] = {0.f, 0.f, 0.f, 0.f};

    const float wfreq = 1.57079632679f / (float)lengths[b];

    for (int s0 = 0; s0 < S_; s0 += KVCHUNK) {
        for (int t = tid; t < KVCHUNK * DH_ / 4; t += 256) {  // 2 iters
            int idx = t * 4;
            int sc = idx >> 5;
            int d  = idx & 31;
            size_t g = ((size_t)b * S_ + s0 + sc) * D_ + h * DH_ + d;
            *(uint2*)&Ks[sc][d] = *(const uint2*)&Kb[g];
            *(uint2*)&Vs[sc][d] = *(const uint2*)&Vb[g];
        }
        if (tid < KVCHUNK) {
            float th = wfreq * (float)(s0 + tid);
            Cs[tid] = __cosf(th);
            Ss[tid] = __sinf(th);
        }
        __syncthreads();

#pragma unroll
        for (int p = 0; p < 4; ++p) {
            int pi = tid + p * 256;
            int d = pi >> 5;
            int e = pi & 31;
            float ac = accc[p], as = accs[p];
            float kc_a = kc_sum[p], ks_a = ks_sum[p];
#pragma unroll 4
            for (int sc = 0; sc < KVCHUNK; ++sc) {
                float kd = bf2f(Ks[sc][d]);
                float ve = bf2f(Vs[sc][e]);
                float kc = kd * Cs[sc];
                float ks = kd * Ss[sc];
                ac += kc * ve;
                as += ks * ve;
                if (e == 0) { kc_a += kc; ks_a += ks; }
            }
            accc[p] = ac; accs[p] = as;
            kc_sum[p] = kc_a; ks_sum[p] = ks_a;
        }
        __syncthreads();
    }

    const size_t bh = (size_t)(b * H_ + h);
#pragma unroll
    for (int p = 0; p < 4; ++p) {
        int pi = tid + p * 256;
        int d = pi >> 5;
        int e = pi & 31;
        kvc[bh * (DH_ * DH_) + d * DH_ + e] = accc[p];
        kvs[bh * (DH_ * DH_) + d * DH_ + e] = accs[p];
        if (e == 0) {
            ksc[bh * DH_ + d] = kc_sum[p];
            kss[bh * DH_ + d] = ks_sum[p];
        }
    }
}

// =================================================================
// attn_num: one workgroup per token; thread t -> (h = t/32, e = t%32)
//   num[e] = sum_d (q*cos)*kv_c[d,e] + (q*sin)*kv_s[d,e]
//   den    = sum_d (q*cos)*ksc[d]    + (q*sin)*kss[d]  (+eps)
//   out    = bf16(num/den)   -> input to Wo GEMM
// =================================================================
__global__ void __launch_bounds__(256)
attn_num(const unsigned short* __restrict__ Qb,
         const float* __restrict__ kvc, const float* __restrict__ kvs,
         const float* __restrict__ ksc, const float* __restrict__ kss,
         const int* __restrict__ lengths,
         unsigned short* __restrict__ Ab)
{
    const int token = blockIdx.x;
    const int b = token / S_;
    const int s = token - b * S_;
    const int tid = threadIdx.x;
    const int h = tid >> 5;
    const int e = tid & 31;

    __shared__ float qsh[D_];
    qsh[tid] = bf2f(Qb[(size_t)token * D_ + tid]);
    __syncthreads();

    float th = 1.57079632679f * (float)s / (float)lengths[b];
    float c = __cosf(th), sn = __sinf(th);

    const size_t bh = (size_t)(b * H_ + h);
    const float* KVc = kvc + bh * (DH_ * DH_);
    const float* KVs = kvs + bh * (DH_ * DH_);
    const float* Kc  = ksc + bh * DH_;
    const float* Kn  = kss + bh * DH_;

    float num = 0.f, den = 0.f;
#pragma unroll 8
    for (int d = 0; d < DH_; ++d) {
        float q  = qsh[h * DH_ + d];
        float qc = q * c, qs = q * sn;
        num += qc * KVc[d * DH_ + e] + qs * KVs[d * DH_ + e];
        den += qc * Kc[d] + qs * Kn[d];
    }
    Ab[(size_t)token * D_ + tid] = f2bf(num / (den + 1e-5f));
}

// =================================================================
// small utility kernels
// =================================================================
__global__ void embed_kernel(const int* __restrict__ ids,
                             const float* __restrict__ emb,
                             float* __restrict__ xf,
                             unsigned short* __restrict__ xb)
{
    const int token = blockIdx.x;
    const int t = threadIdx.x;
    int id = ids[token];
    float v = emb[(size_t)id * D_ + t];
    size_t idx = (size_t)token * D_ + t;
    xf[idx] = v;
    xb[idx] = f2bf(v);
}

__global__ void cvt_f32_bf16(const float* __restrict__ src,
                             unsigned short* __restrict__ dst, int n)
{
    int i = blockIdx.x * 256 + threadIdx.x;
    if (i < n) dst[i] = f2bf(src[i]);
}

__global__ void mask_kernel(const int* __restrict__ am,
                            float* __restrict__ rowmask, int n)
{
    int i = blockIdx.x * 256 + threadIdx.x;
    if (i < n) rowmask[i] = (am[i] > 0) ? 1.0f : 0.0f;
}

// =================================================================
// host-side launcher
// =================================================================
extern "C" void kernel_launch(void* const* d_in, const int* in_sizes, int n_in,
                              void* d_out, int out_size, void* d_ws, size_t ws_size,
                              hipStream_t stream)
{
    (void)in_sizes; (void)n_in; (void)out_size; (void)ws_size;

    const int*   ids   = (const int*)d_in[0];
    const int*   amask = (const int*)d_in[1];
    const int*   lens  = (const int*)d_in[2];
    const float* emb   = (const float*)d_in[3];
    const float* Wq    = (const float*)d_in[4];
    const float* Wk    = (const float*)d_in[5];
    const float* Wv    = (const float*)d_in[6];
    const float* Wo    = (const float*)d_in[7];
    const float* W1    = (const float*)d_in[8];
    const float* W2    = (const float*)d_in[9];
    const float* resw  = (const float*)d_in[10];

    const size_t M = (size_t)M_TOK;

    // ---- workspace layout (256B aligned slices) ----
    char* p = (char*)d_ws;
    auto alloc = [&](size_t bytes) -> char* {
        char* r = p;
        p += (bytes + 255) & ~(size_t)255;
        return r;
    };
    float*          xf32 = (float*)alloc(M * D_ * sizeof(float));
    unsigned short* xbf  = (unsigned short*)alloc(M * D_ * 2);
    // q|k|v|attn during attention phase; whole region reused as FFN hidden
    unsigned short* qkva = (unsigned short*)alloc(M * (size_t)F_ * 2);
    unsigned short* qb = qkva;
    unsigned short* kb = qkva + M * D_;
    unsigned short* vb = qkva + 2 * M * D_;
    unsigned short* ab = qkva + 3 * M * D_;
    unsigned short* hb = qkva;           // FFN hidden [M, F]
    float* rowmask = (float*)alloc(M * sizeof(float));
    float* kvc = (float*)alloc((size_t)B_ * H_ * DH_ * DH_ * sizeof(float));
    float* kvs = (float*)alloc((size_t)B_ * H_ * DH_ * DH_ * sizeof(float));
    float* ksc = (float*)alloc((size_t)B_ * H_ * DH_ * sizeof(float));
    float* kss = (float*)alloc((size_t)B_ * H_ * DH_ * sizeof(float));
    unsigned short* wqb = (unsigned short*)alloc((size_t)L_ * D_ * D_ * 2);
    unsigned short* wkb = (unsigned short*)alloc((size_t)L_ * D_ * D_ * 2);
    unsigned short* wvb = (unsigned short*)alloc((size_t)L_ * D_ * D_ * 2);
    unsigned short* wob = (unsigned short*)alloc((size_t)L_ * D_ * D_ * 2);
    unsigned short* w1b = (unsigned short*)alloc((size_t)L_ * D_ * F_ * 2);
    unsigned short* w2b = (unsigned short*)alloc((size_t)L_ * F_ * D_ * 2);

    // ---- weight conversion to bf16 ----
    {
        int n_dd = L_ * D_ * D_;     // 262144
        int n_df = L_ * D_ * F_;     // 1048576
        cvt_f32_bf16<<<(n_dd + 255) / 256, 256, 0, stream>>>(Wq, wqb, n_dd);
        cvt_f32_bf16<<<(n_dd + 255) / 256, 256, 0, stream>>>(Wk, wkb, n_dd);
        cvt_f32_bf16<<<(n_dd + 255) / 256, 256, 0, stream>>>(Wv, wvb, n_dd);
        cvt_f32_bf16<<<(n_dd + 255) / 256, 256, 0, stream>>>(Wo, wob, n_dd);
        cvt_f32_bf16<<<(n_df + 255) / 256, 256, 0, stream>>>(W1, w1b, n_df);
        cvt_f32_bf16<<<(n_df + 255) / 256, 256, 0, stream>>>(W2, w2b, n_df);
    }
    mask_kernel<<<(int)((M + 255) / 256), 256, 0, stream>>>(amask, rowmask, (int)M);
    embed_kernel<<<(int)M, 256, 0, stream>>>(ids, emb, xf32, xbf);

    const dim3 blk(256);
    const dim3 grid_d(D_ / BN, (unsigned)(M / BM));   // N=256  -> (1, 1024)
    const dim3 grid_f(F_ / BN, (unsigned)(M / BM));   // N=1024 -> (4, 1024)

    for (int l = 0; l < L_; ++l) {
        // Q = relu(x @ Wq)
        gemm_bf16<<<grid_d, blk, 0, stream>>>(xbf, wqb + (size_t)l * D_ * D_,
            (int)M, D_, D_, 1, nullptr, qb, nullptr, nullptr, 0);
        // K = relu(x @ Wk) * mask
        gemm_bf16<<<grid_d, blk, 0, stream>>>(xbf, wkb + (size_t)l * D_ * D_,
            (int)M, D_, D_, 1, rowmask, kb, nullptr, nullptr, 0);
        // V = (x @ Wv) * mask
        gemm_bf16<<<grid_d, blk, 0, stream>>>(xbf, wvb + (size_t)l * D_ * D_,
            (int)M, D_, D_, 0, rowmask, vb, nullptr, nullptr, 0);
        // KV summaries + key sums
        kv_summary<<<B_ * H_, blk, 0, stream>>>(kb, vb, lens, kvc, kvs, ksc, kss);
        // numerator / denominator -> attention output (pre-Wo)
        attn_num<<<(int)M, blk, 0, stream>>>(qb, kvc, kvs, ksc, kss, lens, ab);
        // x += rw[l,0] * (attn @ Wo)   (updates xf32 and xbf)
        gemm_bf16<<<grid_d, blk, 0, stream>>>(ab, wob + (size_t)l * D_ * D_,
            (int)M, D_, D_, 0, nullptr, xbf, xf32, resw, 2 * l);
        // h = relu(x @ W1)
        gemm_bf16<<<grid_f, blk, 0, stream>>>(xbf, w1b + (size_t)l * D_ * F_,
            (int)M, F_, D_, 1, nullptr, hb, nullptr, nullptr, 0);
        // x += rw[l,1] * (h @ W2)
        gemm_bf16<<<grid_d, blk, 0, stream>>>(hb, w2b + (size_t)l * F_ * D_,
            (int)M, D_, F_, 0, nullptr, xbf, xf32, resw, 2 * l + 1);
    }

    hipMemcpyAsync(d_out, xf32, M * D_ * sizeof(float),
                   hipMemcpyDeviceToDevice, stream);
}
